// MultiHeadAttention_87024627352037
// MI455X (gfx1250) — compile-verified, matrix operand order checked by reading, not run on hardware
//
#include <hip/hip_runtime.h>
#include <hip/hip_bf16.h>

// ---------------------------------------------------------------------------
// MHA block for MI455X (gfx1250): bf16 WMMA everywhere, flash-style attention
// with TDM (tensor_load_to_lds) double-buffered K tiles, software-pipelined
// projection GEMMs, transpose-free operand layouts.
// ---------------------------------------------------------------------------

#define B_ 2
#define S_ 2048
#define D_ 1024
#define H_ 16
#define ND_ 64
#define M_ 4096            // B_*S_

typedef __attribute__((ext_vector_type(4)))  __bf16 v4bf;
typedef __attribute__((ext_vector_type(8)))  __bf16 v8bf;
typedef __attribute__((ext_vector_type(16))) __bf16 v16bf;
typedef __attribute__((ext_vector_type(8)))  float  v8f;
typedef __attribute__((ext_vector_type(4)))  unsigned int u32x4;
typedef __attribute__((ext_vector_type(4)))  int i32x4;
typedef __attribute__((ext_vector_type(8)))  int i32x8;

__device__ __forceinline__ __bf16 f2bf(float f) { return (__bf16)f; }

__device__ __forceinline__ v8f v8f_zero() {
  v8f z;
#pragma unroll
  for (int i = 0; i < 8; ++i) z[i] = 0.0f;
  return z;
}

// Load one 32-k-deep A/B fragment slice for this lane.
// Caller passes p = row_base + kstep*32 + (lane>=16 ? 8 : 0).
// Fragment elements 0..7 <- p[0..7], 8..15 <- p[16..23]  (ISA 16-bit A layout).
__device__ __forceinline__ v16bf load_frag(const __bf16* p) {
  v8bf lo = *(const v8bf*)(p);
  v8bf hi = *(const v8bf*)(p + 16);
  return __builtin_shufflevector(lo, hi, 0,1,2,3,4,5,6,7,8,9,10,11,12,13,14,15);
}

__device__ __forceinline__ v8f wmma_bf16(v16bf a, v16bf b, v8f c) {
  return __builtin_amdgcn_wmma_f32_16x16x32_bf16(false, a, false, b,
                                                 (short)0, c, false, false);
}

// ---------------------------------------------------------------------------
// Tensor Data Mover: async DMA of a 64x64 bf16 tile (row stride 64 elems)
// from global memory into LDS. Descriptor per CDNA5 ISA §8 (D# groups 0/1).
// ---------------------------------------------------------------------------
__device__ __forceinline__ void tdm_load_tile_64x64_bf16(const __bf16* gsrc,
                                                         unsigned int lds_byte) {
  const unsigned long long ga = (unsigned long long)(size_t)gsrc;
  u32x4 g0;
  g0[0] = 1u;                                          // count=1 (valid, user)
  g0[1] = lds_byte;                                    // lds_addr
  g0[2] = (unsigned int)(ga & 0xffffffffull);          // global_addr[31:0]
  g0[3] = (unsigned int)((ga >> 32) & 0x01ffffffull)   // global_addr[56:32]
          | 0x80000000u;                               // type=2 ("image")
  i32x8 g1;
  g1[0] = 0x00010000;        // wg_mask=0, data_size=1 (2 bytes)
  g1[1] = (64 << 16);        // atomic_barrier_addr=0 | tensor_dim0[15:0]=64
  g1[2] = (64 << 16);        // tensor_dim0[31:16]=0  | tensor_dim1[15:0]=64
  g1[3] = (64 << 16);        // tensor_dim1[31:16]=0  | tile_dim0=64
  g1[4] = 64;                // tile_dim1=64 | tile_dim2=0
  g1[5] = 64;                // tensor_dim0_stride[31:0]=64
  g1[6] = 0;                 // stride0[47:32]=0 | stride1[15:0]=0
  g1[7] = 0;
  i32x4 z4;
  z4[0] = 0; z4[1] = 0; z4[2] = 0; z4[3] = 0;
#if defined(__clang_major__) && (__clang_major__ >= 23)
  i32x8 z8;
#pragma unroll
  for (int i = 0; i < 8; ++i) z8[i] = 0;
  __builtin_amdgcn_tensor_load_to_lds(g0, g1, z4, z4, z8, 0);
#else
  __builtin_amdgcn_tensor_load_to_lds(g0, g1, z4, z4, 0);
#endif
}

// ---------------------------------------------------------------------------
// 1) f32 -> bf16 conversion (bandwidth pass, ~6us at 23.3 TB/s for all arrays)
// ---------------------------------------------------------------------------
__global__ __launch_bounds__(256) void cvt_bf16_kernel(
    const float* __restrict__ src, __bf16* __restrict__ dst, int n4) {
  int i = blockIdx.x * 256 + threadIdx.x;
  if (i < n4) {
    float4 v = ((const float4*)src)[i];
    v4bf o;
    o[0] = f2bf(v.x); o[1] = f2bf(v.y); o[2] = f2bf(v.z); o[3] = f2bf(v.w);
    ((v4bf*)dst)[i] = o;
  }
}

// ---------------------------------------------------------------------------
// 2) GEMM  out = A(MxK) @ W^T(NxK) + bias   — bf16 WMMA, f32 accum,
//    software-pipelined fragment loads (next k-step issued before consuming
//    current, so s_wait_loadcnt overlaps WMMA issue instead of draining).
//    mode 0/1: write head-split  [B,H,S,ND] bf16   (Q, K)
//    mode 2  : write transposed  [B,H,ND,S] bf16   (V -> contiguous P@V B-frags)
//    mode 3  : write f32 (MxN) with +bias +residual (O projection)
// ---------------------------------------------------------------------------
__global__ __launch_bounds__(128) void gemm_bf16_kernel(
    const __bf16* __restrict__ Abf, const __bf16* __restrict__ Wbf,
    const float* __restrict__ bias, const float* __restrict__ resid,
    __bf16* __restrict__ out_bf, float* __restrict__ out_f, int mode) {
  const int lane = threadIdx.x & 31;
  const int wave = threadIdx.x >> 5;
  const int half = lane >> 4;
  const int l16  = lane & 15;
  const int m0 = blockIdx.y * 64 + wave * 16;
  const int n0 = blockIdx.x * 64;

  v8f acc[4];
#pragma unroll
  for (int t = 0; t < 4; ++t) acc[t] = v8f_zero();

  const __bf16* aptr = Abf + (size_t)(m0 + l16) * D_ + half * 8;
  const __bf16* bp0 = Wbf + (size_t)(n0 +  0 + l16) * D_ + half * 8;
  const __bf16* bp1 = Wbf + (size_t)(n0 + 16 + l16) * D_ + half * 8;
  const __bf16* bp2 = Wbf + (size_t)(n0 + 32 + l16) * D_ + half * 8;
  const __bf16* bp3 = Wbf + (size_t)(n0 + 48 + l16) * D_ + half * 8;

  v16bf af = load_frag(aptr);
  v16bf b0 = load_frag(bp0), b1 = load_frag(bp1);
  v16bf b2 = load_frag(bp2), b3 = load_frag(bp3);

  for (int k0 = 0; k0 < D_; k0 += 32) {
    const int kn = k0 + 32;
    v16bf afn = af, b0n = b0, b1n = b1, b2n = b2, b3n = b3;
    if (kn < D_) {                          // issue next loads before waiting
      afn = load_frag(aptr + kn);
      b0n = load_frag(bp0 + kn);
      b1n = load_frag(bp1 + kn);
      b2n = load_frag(bp2 + kn);
      b3n = load_frag(bp3 + kn);
    }
    acc[0] = wmma_bf16(af, b0, acc[0]);
    acc[1] = wmma_bf16(af, b1, acc[1]);
    acc[2] = wmma_bf16(af, b2, acc[2]);
    acc[3] = wmma_bf16(af, b3, acc[3]);
    af = afn; b0 = b0n; b1 = b1n; b2 = b2n; b3 = b3n;
  }

#pragma unroll
  for (int t = 0; t < 4; ++t) {
#pragma unroll
    for (int r = 0; r < 8; ++r) {
      const int gm = m0 + r + 8 * half;          // C layout: M = r + 8*(lane>>4)
      const int n  = n0 + t * 16 + l16;          //           N = lane&15
      float v = acc[t][r] + bias[n];
      if (mode == 3) {
        out_f[(size_t)gm * D_ + n] = v + resid[(size_t)gm * D_ + n];
      } else {
        const int bidx = gm >> 11;               // / S_
        const int srow = gm & (S_ - 1);
        const int hh = n >> 6, nd = n & 63;
        size_t idx;
        if (mode == 2)
          idx = ((size_t)(bidx * H_ + hh) * ND_ + nd) * S_ + srow;   // V^T
        else
          idx = ((size_t)(bidx * H_ + hh) * S_ + srow) * ND_ + nd;   // Q / K
        out_bf[idx] = f2bf(v);
      }
    }
  }
}

// ---------------------------------------------------------------------------
// 3) Causal flash attention. One wave per (b, h, 16 q-rows).
//    K tiles (64 keys x 64 dims, 8KB) are DMA'd by the Tensor Data Mover into
//    double-buffered LDS (compute block j while TDM fetches j+1), synchronized
//    with s_wait_tensorcnt. Per 64-key block: 8 WMMA (Q.K^T) + 8 WMMA (P.V),
//    online softmax with half-wave shuffles, P via 2KB LDS bounce.
// ---------------------------------------------------------------------------
__global__ __launch_bounds__(32) void attn_kernel(
    const __bf16* __restrict__ Qh, const __bf16* __restrict__ Kh,
    const __bf16* __restrict__ Vt, __bf16* __restrict__ ctx) {
  __shared__ __align__(16) __bf16 Ksh[2][64 * 64];   // TDM double buffer
  __shared__ __align__(16) __bf16 Psh[16 * 64];

  const int lane = threadIdx.x;
  const int half = lane >> 4;
  const int l16  = lane & 15;
  const int b = blockIdx.z, h = blockIdx.y;
  const int q0 = blockIdx.x * 16;

  const __bf16* Qb = Qh + (size_t)(b * H_ + h) * S_ * ND_;   // [S][ND]
  const __bf16* Kb = Kh + (size_t)(b * H_ + h) * S_ * ND_;   // [S][ND]
  const __bf16* Vb = Vt + (size_t)(b * H_ + h) * ND_ * S_;   // [ND][S]

  const unsigned int ksh_lds[2] = {
      (unsigned int)(size_t)&Ksh[0][0],   // generic addr low 32 bits == LDS addr
      (unsigned int)(size_t)&Ksh[1][0]};

  const v16bf qf0 = load_frag(Qb + (size_t)(q0 + l16) * ND_ + half * 8);
  const v16bf qf1 = load_frag(Qb + (size_t)(q0 + l16) * ND_ + 32 + half * 8);

  float mrun[8], lrun[8];
  v8f oacc[4];
#pragma unroll
  for (int r = 0; r < 8; ++r) { mrun[r] = -1e30f; lrun[r] = 0.0f; }
#pragma unroll
  for (int t = 0; t < 4; ++t) oacc[t] = v8f_zero();

  const int jmax = (q0 + 15) >> 6;               // causal key-block bound
  tdm_load_tile_64x64_bf16(Kb, ksh_lds[0]);      // prime the pipeline

  for (int j = 0; j <= jmax; ++j) {
    const int k0 = j << 6;
    if (j < jmax) {                              // TDM prefetch of next K tile
      tdm_load_tile_64x64_bf16(Kb + (size_t)(k0 + 64) * ND_, ksh_lds[(j + 1) & 1]);
      __builtin_amdgcn_s_wait_tensorcnt((short)1);   // oldest (block j) done
      __builtin_prefetch(Vb + (size_t)(l16 * 4) * S_ + k0 + 64, 0, 3);
    } else {
      __builtin_amdgcn_s_wait_tensorcnt((short)0);
    }

    const __bf16* kt = &Ksh[j & 1][0];
    v8f sc[4];
#pragma unroll
    for (int t = 0; t < 4; ++t) sc[t] = v8f_zero();
#pragma unroll
    for (int t = 0; t < 4; ++t) {
      const __bf16* kp = kt + (t * 16 + l16) * 64 + half * 8;   // ds_load_b128
      v16bf kf0 = load_frag(kp);
      v16bf kf1 = load_frag(kp + 32);
      sc[t] = wmma_bf16(qf0, kf0, sc[t]);
      sc[t] = wmma_bf16(qf1, kf1, sc[t]);
    }

    // online softmax over this 16x64 score tile
#pragma unroll
    for (int r = 0; r < 8; ++r) {
      const int qrow = q0 + r + 8 * half;
      float vals[4];
      float rowmax = -1e30f;
#pragma unroll
      for (int t = 0; t < 4; ++t) {
        const int key = k0 + t * 16 + l16;
        float v = sc[t][r] * 0.125f;             // 1/sqrt(64)
        if (key > qrow) v = -1e30f;              // causal mask
        vals[t] = v;
        rowmax = fmaxf(rowmax, v);
      }
      rowmax = fmaxf(rowmax, __shfl_xor(rowmax, 1));
      rowmax = fmaxf(rowmax, __shfl_xor(rowmax, 2));
      rowmax = fmaxf(rowmax, __shfl_xor(rowmax, 4));
      rowmax = fmaxf(rowmax, __shfl_xor(rowmax, 8));   // halves stay separate

      const float mnew  = fmaxf(mrun[r], rowmax);
      const float rescl = __expf(mrun[r] - mnew);
      mrun[r] = mnew;

      float rs = 0.0f;
#pragma unroll
      for (int t = 0; t < 4; ++t) {
        const float e = __expf(vals[t] - mnew);
        rs += e;
        Psh[(r + 8 * half) * 64 + t * 16 + l16] = f2bf(e);
      }
      rs += __shfl_xor(rs, 1);
      rs += __shfl_xor(rs, 2);
      rs += __shfl_xor(rs, 4);
      rs += __shfl_xor(rs, 8);
      lrun[r] = lrun[r] * rescl + rs;
#pragma unroll
      for (int t = 0; t < 4; ++t) oacc[t][r] *= rescl;
    }
    __syncthreads();

    // re-read P in A-fragment layout from LDS, multiply by V^T columns
    const v16bf pf0 = load_frag(&Psh[l16 * 64 + half * 8]);
    const v16bf pf1 = load_frag(&Psh[l16 * 64 + 32 + half * 8]);
#pragma unroll
    for (int t = 0; t < 4; ++t) {
      const __bf16* vp = Vb + (size_t)(t * 16 + l16) * S_ + k0 + half * 8;
      v16bf vf0 = load_frag(vp);
      v16bf vf1 = load_frag(vp + 32);
      oacc[t] = wmma_bf16(pf0, vf0, oacc[t]);
      oacc[t] = wmma_bf16(pf1, vf1, oacc[t]);
    }
    __syncthreads();
  }

  // normalize and write merged-head ctx [B,S,D] bf16
#pragma unroll
  for (int r = 0; r < 8; ++r) {
    const float inv = 1.0f / lrun[r];
    const int srow = q0 + r + 8 * half;
#pragma unroll
    for (int t = 0; t < 4; ++t) {
      ctx[(size_t)(b * S_ + srow) * D_ + h * 64 + t * 16 + l16] =
          f2bf(oacc[t][r] * inv);
    }
  }
}

// ---------------------------------------------------------------------------
// 4) LayerNorm: one 256-thread block per row of 1024.
// ---------------------------------------------------------------------------
__global__ __launch_bounds__(256) void ln_kernel(
    const float* __restrict__ y, const float* __restrict__ gamma,
    const float* __restrict__ beta, float* __restrict__ out) {
  __shared__ float ssum[8], ssq[8];
  const int row = blockIdx.x;
  const int tid = threadIdx.x;
  const float4 v = ((const float4*)(y + (size_t)row * D_))[tid];

  float s = v.x + v.y + v.z + v.w;
  float q = v.x * v.x + v.y * v.y + v.z * v.z + v.w * v.w;
#pragma unroll
  for (int m = 1; m <= 16; m <<= 1) {
    s += __shfl_xor(s, m);
    q += __shfl_xor(q, m);
  }
  const int wid = tid >> 5, lane = tid & 31;
  if (lane == 0) { ssum[wid] = s; ssq[wid] = q; }
  __syncthreads();
  if (wid == 0) {
    float a  = (lane < 8) ? ssum[lane] : 0.0f;
    float bq = (lane < 8) ? ssq[lane] : 0.0f;
#pragma unroll
    for (int m = 1; m <= 4; m <<= 1) {
      a  += __shfl_xor(a, m);
      bq += __shfl_xor(bq, m);
    }
    if (lane == 0) { ssum[0] = a; ssq[0] = bq; }
  }
  __syncthreads();

  const float mu   = ssum[0] * (1.0f / D_);
  const float var  = ssq[0] * (1.0f / D_) - mu * mu;
  const float rinv = rsqrtf(var + 1e-5f);

  const float4 g  = ((const float4*)gamma)[tid];
  const float4 bt = ((const float4*)beta)[tid];
  float4 o;
  o.x = (v.x - mu) * rinv * g.x + bt.x;
  o.y = (v.y - mu) * rinv * g.y + bt.y;
  o.z = (v.z - mu) * rinv * g.z + bt.z;
  o.w = (v.w - mu) * rinv * g.w + bt.w;
  ((float4*)(out + (size_t)row * D_))[tid] = o;
}

// ---------------------------------------------------------------------------
// Host-side orchestration (graph-capture safe: launches only)
// ---------------------------------------------------------------------------
extern "C" void kernel_launch(void* const* d_in, const int* in_sizes, int n_in,
                              void* d_out, int out_size, void* d_ws, size_t ws_size,
                              hipStream_t stream) {
  (void)in_sizes; (void)n_in; (void)out_size; (void)ws_size;
  const float* x_q   = (const float*)d_in[0];
  const float* x_k   = (const float*)d_in[1];
  const float* x_v   = (const float*)d_in[2];
  /* d_in[3] = mask: causality is computed analytically in attn_kernel */
  const float* Wq    = (const float*)d_in[4];
  const float* bq    = (const float*)d_in[5];
  const float* Wk    = (const float*)d_in[6];
  const float* bk    = (const float*)d_in[7];
  const float* Wv    = (const float*)d_in[8];
  const float* bv    = (const float*)d_in[9];
  const float* Wo    = (const float*)d_in[10];
  const float* bo    = (const float*)d_in[11];
  const float* gamma = (const float*)d_in[12];
  const float* beta  = (const float*)d_in[13];
  float* out = (float*)d_out;

  char* ws = (char*)d_ws;
  const size_t nMD = (size_t)M_ * D_;           // 4M elems
  const size_t nDD = (size_t)D_ * D_;           // 1M elems
  size_t off = 0;
  __bf16* xq_bf = (__bf16*)(ws + off); off += nMD * 2;
  __bf16* xk_bf = (__bf16*)(ws + off); off += nMD * 2;
  __bf16* xv_bf = (__bf16*)(ws + off); off += nMD * 2;
  __bf16* wq_bf = (__bf16*)(ws + off); off += nDD * 2;
  __bf16* wk_bf = (__bf16*)(ws + off); off += nDD * 2;
  __bf16* wv_bf = (__bf16*)(ws + off); off += nDD * 2;
  __bf16* wo_bf = (__bf16*)(ws + off); off += nDD * 2;
  __bf16* Qh    = (__bf16*)(ws + off); off += nMD * 2;   // [B,H,S,ND]
  __bf16* Kh    = (__bf16*)(ws + off); off += nMD * 2;   // [B,H,S,ND]
  __bf16* Vt    = (__bf16*)(ws + off); off += nMD * 2;   // [B,H,ND,S]
  __bf16* ctxb  = (__bf16*)(ws + off); off += nMD * 2;   // [B,S,D]
  float*  yf    = (float*)(ws + off);  off += nMD * 4;   // pre-LN f32

  // 1) conversions
  const int n4MD = (int)(nMD / 4), n4DD = (int)(nDD / 4);
  cvt_bf16_kernel<<<(n4MD + 255) / 256, 256, 0, stream>>>(x_q, xq_bf, n4MD);
  cvt_bf16_kernel<<<(n4MD + 255) / 256, 256, 0, stream>>>(x_k, xk_bf, n4MD);
  cvt_bf16_kernel<<<(n4MD + 255) / 256, 256, 0, stream>>>(x_v, xv_bf, n4MD);
  cvt_bf16_kernel<<<(n4DD + 255) / 256, 256, 0, stream>>>(Wq, wq_bf, n4DD);
  cvt_bf16_kernel<<<(n4DD + 255) / 256, 256, 0, stream>>>(Wk, wk_bf, n4DD);
  cvt_bf16_kernel<<<(n4DD + 255) / 256, 256, 0, stream>>>(Wv, wv_bf, n4DD);
  cvt_bf16_kernel<<<(n4DD + 255) / 256, 256, 0, stream>>>(Wo, wo_bf, n4DD);

  // 2) Q/K/V projections (WMMA)
  dim3 gproj(D_ / 64, M_ / 64);
  gemm_bf16_kernel<<<gproj, 128, 0, stream>>>(xq_bf, wq_bf, bq, nullptr, Qh, nullptr, 0);
  gemm_bf16_kernel<<<gproj, 128, 0, stream>>>(xk_bf, wk_bf, bk, nullptr, Kh, nullptr, 1);
  gemm_bf16_kernel<<<gproj, 128, 0, stream>>>(xv_bf, wv_bf, bv, nullptr, Vt, nullptr, 2);

  // 3) causal flash attention (WMMA + TDM)
  attn_kernel<<<dim3(S_ / 16, H_, B_), 32, 0, stream>>>(Qh, Kh, Vt, ctxb);

  // 4) output projection + bias + residual (WMMA)
  gemm_bf16_kernel<<<gproj, 128, 0, stream>>>(ctxb, wo_bf, bo, x_q, nullptr, yf, 3);

  // 5) layernorm
  ln_kernel<<<M_, 256, 0, stream>>>(yf, gamma, beta, out);
}
